// LinearAttention_8143257994019
// MI455X (gfx1250) — compile-verified, matrix operand order checked by reading, not run on hardware
//
#include <hip/hip_runtime.h>
#include <hip/hip_bf16.h>

// MI455X / gfx1250 (CDNA5, wave32) fused linear-attention block.
// 3x [16384x1024x1024] bf16 WMMA GEMMs dominate (~103 GFLOP); small batched
// GEMMs (kv, out) also on WMMA. Async global->LDS (ASYNCcnt) double-buffering
// feeds the bf16 stages; fp32 epilogues (GELU/RoPE/z/LayerNorm).
//
// Workspace (~193 MB): qbf/qr/kr/vbf bf16 [8,2048,1024] (32 MB each),
// y f32 (64 MB), ksum f32 [8,1024] (zeroed per call), kvT bf16 [128,64,64].

typedef __attribute__((ext_vector_type(16))) __bf16 v16bf;
typedef __attribute__((ext_vector_type(8)))  float  v8f;

#define BATCH 8
#define SEQ   2048
#define CH    1024
#define HEADS 16
#define DH    64

union V16U { uint4 u[2]; v16bf v; };

__device__ __forceinline__ v8f wmma_bf16(v16bf a, v16bf b, v8f c) {
  return __builtin_amdgcn_wmma_f32_16x16x32_bf16(false, a, false, b, (short)0, c,
                                                 false, false);
}

// A-fragment (16-bit A 16x32, ISA 7.12.2): per lane the 16 K values are two
// contiguous runs of 8: [half*8 .. +7] and [16+half*8 .. +7].
__device__ __forceinline__ int kmap(int e, int half) {
  return (e < 8) ? (half * 8 + e) : (8 + half * 8 + e);
}

// Load a 16-element bf16 fragment row: two 16B chunks at +half*8 and +16+half*8.
__device__ __forceinline__ v16bf frag_rows16(const __bf16* rowp, int half) {
  V16U t;
  const __bf16* p = rowp + half * 8;
  t.u[0] = *(const uint4*)p;
  t.u[1] = *(const uint4*)(p + 16);
  return t.v;
}

__device__ __forceinline__ float gelu_exact(float v) {
  return 0.5f * v * (1.0f + erff(v * 0.70710678118654752f));
}

// ---- CDNA5 async global->LDS (ASYNCcnt) with compile-safe fallback ---------
// Builtin signature (from hipcc diagnostic): param0 is int4* in AS1 (__device__).
typedef int v4i_t __attribute__((vector_size(16)));
typedef __attribute__((address_space(1))) v4i_t* as1_v4i_p;
typedef __attribute__((address_space(3))) v4i_t* as3_v4i_p;

#if __has_builtin(__builtin_amdgcn_global_load_async_to_lds_b128)
#define HAVE_ASYNC_LDS 1
__device__ __forceinline__ void async_copy16(const void* g, void* l) {
  __builtin_amdgcn_global_load_async_to_lds_b128(
      (as1_v4i_p)g, (as3_v4i_p)l, 0, 0);
}
#else
#define HAVE_ASYNC_LDS 0
__device__ __forceinline__ void async_copy16(const void* g, void* l) {
  *(uint4*)l = *(const uint4*)g;   // synchronous fallback
}
#endif

__device__ __forceinline__ void wait_async0() {
#if __has_builtin(__builtin_amdgcn_s_wait_asynccnt)
  __builtin_amdgcn_s_wait_asynccnt(0);
#elif HAVE_ASYNC_LDS
  asm volatile("s_wait_asynccnt 0x0" ::: "memory");
#endif
}

// ---------------------------------------------------------------------------
// Kernel 1: fused QKV projection + bias + GELU(+0.21) + RoPE + k-sum.
// 256 thr = 8 waves; tile M=64 x N=32; waves 4(M) x 2(N); K-step 32.
// ---------------------------------------------------------------------------
__global__ __launch_bounds__(256) void qkv_rope_kernel(
    const float* __restrict__ x,
    const float* __restrict__ Wq, const float* __restrict__ bq,
    const float* __restrict__ Wk, const float* __restrict__ bk,
    const float* __restrict__ Wv, const float* __restrict__ bv,
    __bf16* __restrict__ qbf, __bf16* __restrict__ qr,
    __bf16* __restrict__ kr,  __bf16* __restrict__ vbf,
    float* __restrict__ ksum)
{
  const int tid  = threadIdx.x;
  const int lane = tid & 31;
  const int w    = tid >> 5;
  const int wm   = w & 3;
  const int wn   = w >> 2;
  const int Nbase = blockIdx.x * 32;
  const int Mbase = blockIdx.y * 64;
  const int batch = Mbase >> 11;
  const int nloc  = Mbase & (SEQ - 1);

  __shared__ __bf16 xs[64][40];      // row stride 80B (16B-aligned, conflict-free)
  __shared__ __bf16 wsh[3][32][40];
  __shared__ float  ksum_s[32];

  if (tid < 32) ksum_s[tid] = 0.0f;

  v8f qa = {}; v8f ka = {}; v8f va = {};

  const int xrow = tid >> 2, xcp = (tid & 3) * 8;   // 8 fp32 per thread
  const int wrow = tid >> 3, wcp = (tid & 7) * 4;   // 4 fp32 per thread per W

  for (int k0 = 0; k0 < CH; k0 += 32) {
    __syncthreads();
    // stage x tile 64x32: 2x b128 load -> cvt -> b128 LDS store
    {
      const float* xp = x + (size_t)(Mbase + xrow) * CH + k0 + xcp;
      float4 f0 = *(const float4*)xp;
      float4 f1 = *(const float4*)(xp + 4);
      if (k0 + 32 < CH) __builtin_prefetch(xp + 32, 0, 1);
      __bf16 t[8] = {(__bf16)f0.x, (__bf16)f0.y, (__bf16)f0.z, (__bf16)f0.w,
                     (__bf16)f1.x, (__bf16)f1.y, (__bf16)f1.z, (__bf16)f1.w};
      *(uint4*)&xs[xrow][xcp] = *(const uint4*)t;
    }
    // stage W tiles (B^T rows = W rows, row-major direct)
#pragma unroll
    for (int m = 0; m < 3; ++m) {
      const float* Wp = (m == 0 ? Wq : (m == 1 ? Wk : Wv));
      float4 f = *(const float4*)(Wp + (size_t)(Nbase + wrow) * CH + k0 + wcp);
      __bf16 t[4] = {(__bf16)f.x, (__bf16)f.y, (__bf16)f.z, (__bf16)f.w};
      *(uint2*)&wsh[m][wrow][wcp] = *(const uint2*)t;
    }
    __syncthreads();

    const int r16 = lane & 15, half = lane >> 4;
    v16bf af  = frag_rows16(&xs[wm * 16 + r16][0], half);
    v16bf bqf = frag_rows16(&wsh[0][wn * 16 + r16][0], half);
    v16bf bkf = frag_rows16(&wsh[1][wn * 16 + r16][0], half);
    v16bf bvf = frag_rows16(&wsh[2][wn * 16 + r16][0], half);
    qa = wmma_bf16(af, bqf, qa);
    ka = wmma_bf16(af, bkf, ka);
    va = wmma_bf16(af, bvf, va);
  }

  // ---- epilogue -----------------------------------------------------------
  const int r16 = lane & 15, half = lane >> 4;
  const int c = Nbase + wn * 16 + r16;
  const float bqv = bq[c], bkv = bk[c], bvv = bv[c];

  float qv[8], kv8[8], vv[8];
  float kpart = 0.0f;
#pragma unroll
  for (int r = 0; r < 8; ++r) {
    float qg = gelu_exact(qa[r] + bqv) + 0.21f;
    float kg = gelu_exact(ka[r] + bkv) + 0.21f;
    qv[r] = qg; kv8[r] = kg; vv[r] = va[r] + bvv;
    kpart += kg;
  }
  kpart += __shfl_xor(kpart, 16, 32);
  if (lane < 16) atomicAdd(&ksum_s[wn * 16 + lane], kpart);

  // RoPE (interleaved pairs over full C): pair = adjacent lanes
  const float invf = __expf(-(float)(c & ~1) * (9.210340371976184f / (float)CH));
  const float sgn  = (c & 1) ? 1.0f : -1.0f;
#pragma unroll
  for (int r = 0; r < 8; ++r) {
    int m = wm * 16 + half * 8 + r;
    float s, co;
    __sincosf((float)(nloc + m) * invf, &s, &co);
    float qp = __shfl_xor(qv[r], 1, 32);
    float kp = __shfl_xor(kv8[r], 1, 32);
    float qrv = qv[r] * co + sgn * qp * s;
    float krv = kv8[r] * co + sgn * kp * s;
    size_t idx = (size_t)(Mbase + m) * CH + c;
    qbf[idx] = (__bf16)qv[r];
    qr[idx]  = (__bf16)qrv;
    kr[idx]  = (__bf16)krv;
    vbf[idx] = (__bf16)vv[r];
  }

  __syncthreads();
  if (tid < 32) atomicAdd(&ksum[batch * CH + Nbase + tid], ksum_s[tid]);
}

// ---------------------------------------------------------------------------
// Kernel 2: kv[b,h] = (1/n) * Kr^T V (64x64, K=2048), stored transposed [e][d].
// Async double-buffered LDS staging; 8 waves x 2 subtiles.
// ---------------------------------------------------------------------------
__global__ __launch_bounds__(256) void kv_kernel(
    const __bf16* __restrict__ kr, const __bf16* __restrict__ vbf,
    __bf16* __restrict__ kvT)
{
  const int bh = blockIdx.x;
  const int b = bh >> 4, h = bh & 15;
  const int tid = threadIdx.x, lane = tid & 31, w = tid >> 5;
  const int sm = w >> 1;
  const int sn0 = (w & 1) * 2;

  __shared__ __bf16 ks[2][32][72];   // row stride 144B (16B-aligned)
  __shared__ __bf16 vs[2][32][72];

  const size_t base = ((size_t)b * SEQ) * CH + h * DH;
  const int srow = tid >> 3, sseg = (tid & 7) * 8;   // 16B chunk per thread

  // issue tile 0 into buffer 0
  {
    size_t g = base + (size_t)srow * CH + sseg;
    async_copy16(kr + g,  &ks[0][srow][sseg]);
    async_copy16(vbf + g, &vs[0][srow][sseg]);
  }

  v8f a0 = {}, a1 = {};
  int cur = 0;
  const int r16 = lane & 15, half = lane >> 4;

  for (int it = 0; it < SEQ / 32; ++it) {
    wait_async0();        // drain own ASYNCcnt BEFORE signaling the barrier
    __syncthreads();
    if (it + 1 < SEQ / 32) {       // prefetch next tile into the other buffer
      size_t g = base + (size_t)((it + 1) * 32 + srow) * CH + sseg;
      async_copy16(kr + g,  &ks[cur ^ 1][srow][sseg]);
      async_copy16(vbf + g, &vs[cur ^ 1][srow][sseg]);
    }
    v16bf af, b0, b1;
#pragma unroll
    for (int e = 0; e < 16; ++e) {
      int kk = kmap(e, half);                     // n within 32-chunk
      af[e] = ks[cur][kk][sm * 16 + r16];         // A = Kr^T (transposed LDS read)
      b0[e] = vs[cur][kk][sn0 * 16 + r16];        // B^T[e][n] = V[n][e]
      b1[e] = vs[cur][kk][(sn0 + 1) * 16 + r16];
    }
    a0 = wmma_bf16(af, b0, a0);
    a1 = wmma_bf16(af, b1, a1);
    cur ^= 1;
  }

  const float sc = 1.0f / (float)SEQ;
  __bf16* out = kvT + (size_t)bh * (DH * DH);
#pragma unroll
  for (int r = 0; r < 8; ++r) {
    int d = sm * 16 + half * 8 + r;
    out[(sn0 * 16 + r16) * DH + d]       = (__bf16)(a0[r] * sc);
    out[((sn0 + 1) * 16 + r16) * DH + d] = (__bf16)(a1[r] * sc);
  }
}

// ---------------------------------------------------------------------------
// Kernel 3: y[b,n,h,:] = (qr @ kv) * z,  z = 1/(q . kmean + 1e-6)
// kv staged once via async->LDS; A-fragments straight from global (2x b128).
// ---------------------------------------------------------------------------
__global__ __launch_bounds__(256) void out_kernel(
    const __bf16* __restrict__ qr, const __bf16* __restrict__ qbf,
    const __bf16* __restrict__ kvT, const float* __restrict__ ksum,
    float* __restrict__ y)
{
  const int bh = blockIdx.y;
  const int b = bh >> 4, h = bh & 15;
  const int tid = threadIdx.x, lane = tid & 31, w = tid >> 5;
  const int r16 = lane & 15, half = lane >> 4;
  const int rowbase = blockIdx.x * 128 + w * 16;

  __shared__ __bf16 kv_s[DH][72];    // kvT [e][d], stride 144B
  __shared__ float  kmean_s[DH];

#pragma unroll
  for (int j = 0; j < 2; ++j) {      // 512 x 16B chunks, 2 per thread
    int cc = j * 256 + tid;
    int e = cc >> 3, seg = (cc & 7) * 8;
    async_copy16(kvT + (size_t)bh * (DH * DH) + e * DH + seg, &kv_s[e][seg]);
  }
  if (tid < DH) kmean_s[tid] = ksum[b * CH + h * DH + tid] * (1.0f / (float)SEQ);
  wait_async0();
  __syncthreads();

  const size_t base = ((size_t)b * SEQ) * CH + h * DH;

  v8f acc[4] = {{}, {}, {}, {}};
#pragma unroll
  for (int k0 = 0; k0 < DH; k0 += 32) {
    v16bf af = frag_rows16(qr + base + (size_t)(rowbase + r16) * CH + k0, half);
#pragma unroll
    for (int sn = 0; sn < 4; ++sn) {
      v16bf bf = frag_rows16(&kv_s[sn * 16 + r16][k0], half);
      acc[sn] = wmma_bf16(af, bf, acc[sn]);
    }
  }

  // z per row: half-wave split dot, combine with shfl_xor(16)
  const __bf16* qrow = qbf + base + (size_t)(rowbase + r16) * CH;
  float part = 0.0f;
#pragma unroll
  for (int d = 0; d < 32; ++d) {
    int dd = half * 32 + d;
    part += (float)qrow[dd] * kmean_s[dd];
  }
  part += __shfl_xor(part, 16, 32);
  const float zr16 = 1.0f / (part + 1e-6f);

#pragma unroll
  for (int r = 0; r < 8; ++r) {
    float z = __shfl(zr16, half * 8 + r, 32);
    int row = rowbase + half * 8 + r;
#pragma unroll
    for (int sn = 0; sn < 4; ++sn) {
      y[base + (size_t)row * CH + sn * 16 + r16] = acc[sn][r] * z;
    }
  }
}

// ---------------------------------------------------------------------------
// Kernel 4: out = LayerNorm(y + x) * gamma + beta (eps 1e-12, biased var)
// ---------------------------------------------------------------------------
__global__ __launch_bounds__(256) void ln_kernel(
    const float* __restrict__ y, const float* __restrict__ x,
    const float* __restrict__ gamma, const float* __restrict__ beta,
    float* __restrict__ out)
{
  const int row = blockIdx.x;
  const int tid = threadIdx.x;
  const size_t base = (size_t)row * CH + tid * 4;

  float4 yv = *(const float4*)(y + base);
  float4 xv = *(const float4*)(x + base);
  float4 gv = *(const float4*)(gamma + tid * 4);
  float4 bv = *(const float4*)(beta + tid * 4);

  float v0 = yv.x + xv.x, v1 = yv.y + xv.y, v2 = yv.z + xv.z, v3 = yv.w + xv.w;
  float s  = v0 + v1 + v2 + v3;
  float ss = v0 * v0 + v1 * v1 + v2 * v2 + v3 * v3;

#pragma unroll
  for (int o = 16; o > 0; o >>= 1) {
    s  += __shfl_xor(s, o, 32);
    ss += __shfl_xor(ss, o, 32);
  }
  __shared__ float rs[8], rss[8];
  const int w = tid >> 5, lane = tid & 31;
  if (lane == 0) { rs[w] = s; rss[w] = ss; }
  __syncthreads();
  if (w == 0) {
    s  = (lane < 8) ? rs[lane]  : 0.0f;
    ss = (lane < 8) ? rss[lane] : 0.0f;
#pragma unroll
    for (int o = 4; o > 0; o >>= 1) {
      s  += __shfl_xor(s, o, 32);
      ss += __shfl_xor(ss, o, 32);
    }
    if (lane == 0) { rs[0] = s; rss[0] = ss; }
  }
  __syncthreads();
  const float mean = rs[0] * (1.0f / (float)CH);
  const float var  = rss[0] * (1.0f / (float)CH) - mean * mean;
  const float inv  = rsqrtf(var + 1e-12f);

  float4 o4;
  o4.x = (v0 - mean) * inv * gv.x + bv.x;
  o4.y = (v1 - mean) * inv * gv.y + bv.y;
  o4.z = (v2 - mean) * inv * gv.z + bv.z;
  o4.w = (v3 - mean) * inv * gv.w + bv.w;
  *(float4*)(out + base) = o4;
}

// ---------------------------------------------------------------------------
extern "C" void kernel_launch(void* const* d_in, const int* in_sizes, int n_in,
                              void* d_out, int out_size, void* d_ws, size_t ws_size,
                              hipStream_t stream)
{
  const float* x     = (const float*)d_in[0];
  const float* Wq    = (const float*)d_in[1];
  const float* bq    = (const float*)d_in[2];
  const float* Wk    = (const float*)d_in[3];
  const float* bk    = (const float*)d_in[4];
  const float* Wv    = (const float*)d_in[5];
  const float* bv    = (const float*)d_in[6];
  const float* gamma = (const float*)d_in[7];
  const float* beta  = (const float*)d_in[8];
  float* out = (float*)d_out;

  const size_t E = (size_t)BATCH * SEQ * CH;
  char* p = (char*)d_ws;
  __bf16* qbf = (__bf16*)p; p += E * 2;
  __bf16* qr  = (__bf16*)p; p += E * 2;
  __bf16* kr  = (__bf16*)p; p += E * 2;
  __bf16* vbf = (__bf16*)p; p += E * 2;
  float*  y   = (float*)p;  p += E * 4;
  float*  ksum= (float*)p;  p += (size_t)BATCH * CH * 4;
  __bf16* kvT = (__bf16*)p; p += (size_t)BATCH * HEADS * DH * DH * 2;

  (void)hipMemsetAsync(ksum, 0, (size_t)BATCH * CH * 4, stream);

  qkv_rope_kernel<<<dim3(CH / 32, (BATCH * SEQ) / 64), 256, 0, stream>>>(
      x, Wq, bq, Wk, bk, Wv, bv, qbf, qr, kr, vbf, ksum);

  kv_kernel<<<dim3(BATCH * HEADS), 256, 0, stream>>>(kr, vbf, kvT);

  out_kernel<<<dim3(SEQ / 128, BATCH * HEADS), 256, 0, stream>>>(qr, qbf, kvT, ksum, y);

  ln_kernel<<<dim3(BATCH * SEQ), 256, 0, stream>>>(y, x, gamma, beta, out);
}
